// MultiHeadAttention_23476291240791
// MI455X (gfx1250) — compile-verified, matrix operand order checked by reading, not run on hardware
//
#include <hip/hip_runtime.h>
#include <hip/hip_bf16.h>

typedef __bf16 bf16_t;
typedef __attribute__((ext_vector_type(4)))  __bf16 v4bf;
typedef __attribute__((ext_vector_type(8)))  __bf16 v8bf;
typedef __attribute__((ext_vector_type(16))) __bf16 v16bf;
typedef __attribute__((ext_vector_type(8)))  float  v8f;
typedef __attribute__((ext_vector_type(4)))  int    v4i;

#define S_LEN  2048
#define EMB    1024
#define NH     16
#define HD     64
#define BATCH  2
#define M_ROWS (BATCH * S_LEN)   /* 4096 */
#define N_QKV  (3 * EMB)         /* 3072 */

union Frag {
  v16bf f;
  struct { v8bf lo, hi; } h;
};

__device__ __forceinline__ v8f wmma_bf16(const Frag& a, const Frag& b, v8f c) {
  return __builtin_amdgcn_wmma_f32_16x16x32_bf16(
      false, a.f, false, b.f, (short)0, c, false, false);
}

// ---- CDNA5 async global->LDS staging (ASYNCcnt) with safe fallback --------
#if __has_builtin(__builtin_amdgcn_global_load_async_to_lds_b128)
#define HAVE_ASYNC_LDS 1
#else
#define HAVE_ASYNC_LDS 0
#endif

typedef v4i __attribute__((address_space(1))) gv4i;  // global 128-bit chunk
typedef v4i __attribute__((address_space(3))) lv4i;  // LDS    128-bit chunk

__device__ __forceinline__ void async_g2l_b128(const bf16_t* g, bf16_t* l) {
#if HAVE_ASYNC_LDS
  __builtin_amdgcn_global_load_async_to_lds_b128(
      (gv4i*)(void*)g, (lv4i*)(void*)l, 0, 0);
#else
  *(v8bf*)l = *(const v8bf*)g;   // sync fallback (ds_store path)
#endif
}

#if __has_builtin(__builtin_amdgcn_s_wait_asynccnt)
#define WAIT_ASYNC(n) __builtin_amdgcn_s_wait_asynccnt(n)
#else
#define WAIT_ASYNC(n) asm volatile("s_wait_asynccnt %0" ::"i"(n) : "memory")
#endif

// ---------------------------------------------------------------------------
// fp32 -> bf16 conversion (vectorized 4-wide, grid-stride)
// ---------------------------------------------------------------------------
__global__ void mha_cvt_bf16(const float* __restrict__ src,
                             bf16_t* __restrict__ dst, int n4) {
  for (int i = blockIdx.x * blockDim.x + threadIdx.x; i < n4;
       i += gridDim.x * blockDim.x) {
    float4 v = ((const float4*)src)[i];
    v4bf o;
    o[0] = (bf16_t)v.x; o[1] = (bf16_t)v.y;
    o[2] = (bf16_t)v.z; o[3] = (bf16_t)v.w;
    ((v4bf*)dst)[i] = o;
  }
}

// ---------------------------------------------------------------------------
// QKV projection: C[m,n] = sum_k A[m,k]*W[n,k] (+bias), scatter to Q/K/Vt.
// Wave = 16(M) x 32(N) strip (2 WMMAs/K-step sharing the A fragment).
// Block = 8 waves (4M x 2N) -> 64 x 64 tile.
// ---------------------------------------------------------------------------
__global__ __launch_bounds__(256)
void mha_gemm_qkv(const bf16_t* __restrict__ A, const bf16_t* __restrict__ W,
                  const float* __restrict__ bias,
                  bf16_t* __restrict__ Qb, bf16_t* __restrict__ Kb,
                  bf16_t* __restrict__ Vt) {
  const int lane = threadIdx.x & 31;
  const int wv   = threadIdx.x >> 5;
  const int r    = lane & 15;
  const int half = lane >> 4;
  const int kb   = half * 8;
  const int tileM = blockIdx.x * 64 + (wv & 3) * 16;
  const int tileN = blockIdx.y * 64 + (wv >> 2) * 32;

  const bf16_t* ap  = A + (size_t)(tileM + r) * EMB + kb;
  const bf16_t* wp0 = W + (size_t)(tileN + r) * EMB + kb;
  const bf16_t* wp1 = wp0 + (size_t)16 * EMB;

  v8f zero = {};
  v8f cc[2] = {zero, zero};
  for (int k0 = 0; k0 < EMB; k0 += 32) {
    Frag a, b0, b1;
    a.h.lo  = *(const v8bf*)(ap + k0);
    a.h.hi  = *(const v8bf*)(ap + k0 + 16);
    b0.h.lo = *(const v8bf*)(wp0 + k0);
    b0.h.hi = *(const v8bf*)(wp0 + k0 + 16);
    b1.h.lo = *(const v8bf*)(wp1 + k0);
    b1.h.hi = *(const v8bf*)(wp1 + k0 + 16);
    cc[0] = wmma_bf16(a, b0, cc[0]);
    cc[1] = wmma_bf16(a, b1, cc[1]);
  }

#pragma unroll
  for (int tt = 0; tt < 2; ++tt) {
    const int n  = tileN + tt * 16 + r;
    const float bn = bias[n];
    const int hh = n / (3 * HD);
    const int rr = n % (3 * HD);
#pragma unroll
    for (int v = 0; v < 8; ++v) {
      const int   m   = tileM + v + 8 * half;
      const float val = cc[tt][v] + bn;
      const int   bb  = m >> 11;
      const int   ss  = m & (S_LEN - 1);
      if (rr < HD) {            // Q (fold 1/sqrt(D) = 1/8)
        Qb[(((size_t)bb * NH + hh) * S_LEN + ss) * HD + rr] =
            (bf16_t)(val * 0.125f);
      } else if (rr < 2 * HD) { // K
        Kb[(((size_t)bb * NH + hh) * S_LEN + ss) * HD + (rr - HD)] = (bf16_t)val;
      } else {                  // V stored transposed [b,h,d,s]
        Vt[(((size_t)bb * NH + hh) * HD + (rr - 2 * HD)) * S_LEN + ss] =
            (bf16_t)val;
      }
    }
  }
}

// ---------------------------------------------------------------------------
// Flash attention. 8 waves/block share one (b,h); K/V tiles are staged into
// LDS once per block via async-to-LDS (double buffered, ASYNCcnt), cutting
// cache traffic 8x. Each wave owns a 16-query tile; 32 keys per iteration.
// ---------------------------------------------------------------------------
__global__ __launch_bounds__(256)
void mha_attn(const bf16_t* __restrict__ Qb, const bf16_t* __restrict__ Kb,
              const bf16_t* __restrict__ Vt, bf16_t* __restrict__ Ctx) {
  __shared__ bf16_t ldsK[2][32 * 64];   // K tile: rows=key, cols=d
  __shared__ bf16_t ldsV[2][64 * 32];   // V^T tile: rows=d, cols=key
  __shared__ bf16_t ldsP[8][16 * 32];   // per-wave P relayout

  const int tid  = threadIdx.x;
  const int lane = tid & 31;
  const int wv   = tid >> 5;
  const int r    = lane & 15;
  const int half = lane >> 4;
  const int kb   = half * 8;

  const int gw = blockIdx.x * 8 + wv;   // 0..4095
  const int qt = gw & 127;
  const int bh = gw >> 7;               // all 8 waves of a block share bh

  const bf16_t* q  = Qb + ((size_t)bh * S_LEN + qt * 16) * HD;
  const bf16_t* kk = Kb + (size_t)bh * S_LEN * HD;
  const bf16_t* vt = Vt + (size_t)bh * HD * S_LEN;

  // cooperative staging addresses (256 threads, 16B each)
  const int krow = tid >> 3, kcol = (tid & 7) * 8;   // 32 x 64
  const int vrow = tid >> 2, vcol = (tid & 3) * 8;   // 64 x 32

  // Q fragments: d = 0..31 and 32..63
  Frag aq0, aq1;
  aq0.h.lo = *(const v8bf*)(q + r * HD + kb);
  aq0.h.hi = *(const v8bf*)(q + r * HD + kb + 16);
  aq1.h.lo = *(const v8bf*)(q + r * HD + 32 + kb);
  aq1.h.hi = *(const v8bf*)(q + r * HD + 32 + kb + 16);

  float mrow[8], lrow[8];
  v8f zero = {};
  v8f ctx[4] = {zero, zero, zero, zero};
#pragma unroll
  for (int v = 0; v < 8; ++v) { mrow[v] = -1e30f; lrow[v] = 0.f; }

  // prologue: stage tile 0 into buffer 0 (2 async ops per wave)
  async_g2l_b128(kk + (size_t)krow * HD + kcol, &ldsK[0][krow * 64 + kcol]);
  async_g2l_b128(vt + (size_t)vrow * S_LEN + vcol, &ldsV[0][vrow * 32 + vcol]);

  const int NIT = S_LEN / 32;           // 64
  for (int it = 0; it < NIT; ++it) {
    const int cur = it & 1;
    if (it + 1 < NIT) {
      const int j1 = (it + 1) * 32;
      async_g2l_b128(kk + (size_t)krow * HD + j1 * HD + kcol,
                     &ldsK[1 - cur][krow * 64 + kcol]);
      async_g2l_b128(vt + (size_t)vrow * S_LEN + j1 + vcol,
                     &ldsV[1 - cur][vrow * 32 + vcol]);
      WAIT_ASYNC(2);                    // oldest 2 (current buffer) done
    } else {
      WAIT_ASYNC(0);
    }
    __syncthreads();                    // all waves' staging visible

    const bf16_t* lk = &ldsK[cur][0];
    const bf16_t* lv = &ldsV[cur][0];

    // ---- scores: two 16-key sub-tiles, K-dim = D = 64 ----
    v8f s[2];
#pragma unroll
    for (int t = 0; t < 2; ++t) {
      const bf16_t* kp = lk + (t * 16 + r) * 64;
      Frag b0, b1;
      b0.h.lo = *(const v8bf*)(kp + kb);
      b0.h.hi = *(const v8bf*)(kp + kb + 16);
      b1.h.lo = *(const v8bf*)(kp + 32 + kb);
      b1.h.hi = *(const v8bf*)(kp + 32 + kb + 16);
      v8f acc = zero;
      acc  = wmma_bf16(aq0, b0, acc);
      acc  = wmma_bf16(aq1, b1, acc);
      s[t] = acc;
    }

    // ---- online softmax; lanes 0-15 own rows v, lanes 16-31 rows v+8 ----
#pragma unroll
    for (int v = 0; v < 8; ++v) {
      float t0 = s[0][v], t1 = s[1][v];
      float mx = fmaxf(t0, t1);
#pragma unroll
      for (int off = 1; off < 16; off <<= 1)
        mx = fmaxf(mx, __shfl_xor(mx, off, 32));
      const float mnew  = fmaxf(mrow[v], mx);
      const float alpha = __expf(mrow[v] - mnew);
      const float p0 = __expf(t0 - mnew);
      const float p1 = __expf(t1 - mnew);
      float rs = p0 + p1;
#pragma unroll
      for (int off = 1; off < 16; off <<= 1)
        rs += __shfl_xor(rs, off, 32);
      mrow[v] = mnew;
      lrow[v] = lrow[v] * alpha + rs;
#pragma unroll
      for (int dt = 0; dt < 4; ++dt) ctx[dt][v] *= alpha;
      const int row = v + 8 * half;
      ldsP[wv][row * 32 + r]      = (bf16_t)p0;
      ldsP[wv][row * 32 + 16 + r] = (bf16_t)p1;
    }

    // wave-local LDS fence before cross-lane A-frag reads
    __builtin_amdgcn_wave_barrier();
    asm volatile("s_wait_dscnt 0" ::: "memory");
    __builtin_amdgcn_wave_barrier();

    Frag pa;
    pa.h.lo = *(const v8bf*)&ldsP[wv][r * 32 + kb];
    pa.h.hi = *(const v8bf*)&ldsP[wv][r * 32 + kb + 16];

    // ---- ctx += P[16x32] * V[32x64] from the staged V^T tile ----
#pragma unroll
    for (int dt = 0; dt < 4; ++dt) {
      const bf16_t* vp = lv + (dt * 16 + r) * 32;
      Frag bv;
      bv.h.lo = *(const v8bf*)(vp + kb);
      bv.h.hi = *(const v8bf*)(vp + kb + 16);
      ctx[dt] = wmma_bf16(pa, bv, ctx[dt]);
    }

    __syncthreads();   // everyone done with buf[cur] before it is overwritten
  }

  // ---- normalize and store context: [B,S,H,D] ----
  const int bb = bh >> 4, hh = bh & 15;
#pragma unroll
  for (int v = 0; v < 8; ++v) {
    const float inv = 1.0f / lrow[v];
    const int   row = v + 8 * half;
    const int   ss  = qt * 16 + row;
#pragma unroll
    for (int dt = 0; dt < 4; ++dt)
      Ctx[(((size_t)bb * S_LEN + ss) * NH + hh) * HD + dt * 16 + r] =
          (bf16_t)(ctx[dt][v] * inv);
  }
}

// ---------------------------------------------------------------------------
// Output projection: out[m,n] = sum_k Ctx[m,k]*W_out[n,k] + b_out[n]  (fp32)
// Same 16x32-per-wave structure as the QKV GEMM.
// ---------------------------------------------------------------------------
__global__ __launch_bounds__(256)
void mha_gemm_out(const bf16_t* __restrict__ A, const bf16_t* __restrict__ W,
                  const float* __restrict__ bias, float* __restrict__ out) {
  const int lane = threadIdx.x & 31;
  const int wv   = threadIdx.x >> 5;
  const int r    = lane & 15;
  const int half = lane >> 4;
  const int kb   = half * 8;
  const int tileM = blockIdx.x * 64 + (wv & 3) * 16;
  const int tileN = blockIdx.y * 64 + (wv >> 2) * 32;

  const bf16_t* ap  = A + (size_t)(tileM + r) * EMB + kb;
  const bf16_t* wp0 = W + (size_t)(tileN + r) * EMB + kb;
  const bf16_t* wp1 = wp0 + (size_t)16 * EMB;

  v8f zero = {};
  v8f cc[2] = {zero, zero};
  for (int k0 = 0; k0 < EMB; k0 += 32) {
    Frag a, b0, b1;
    a.h.lo  = *(const v8bf*)(ap + k0);
    a.h.hi  = *(const v8bf*)(ap + k0 + 16);
    b0.h.lo = *(const v8bf*)(wp0 + k0);
    b0.h.hi = *(const v8bf*)(wp0 + k0 + 16);
    b1.h.lo = *(const v8bf*)(wp1 + k0);
    b1.h.hi = *(const v8bf*)(wp1 + k0 + 16);
    cc[0] = wmma_bf16(a, b0, cc[0]);
    cc[1] = wmma_bf16(a, b1, cc[1]);
  }

#pragma unroll
  for (int tt = 0; tt < 2; ++tt) {
    const int n = tileN + tt * 16 + r;
    const float bn = bias[n];
#pragma unroll
    for (int v = 0; v < 8; ++v) {
      const int m = tileM + v + 8 * half;
      out[(size_t)m * EMB + n] = cc[tt][v] + bn;
    }
  }
}

// ---------------------------------------------------------------------------
extern "C" void kernel_launch(void* const* d_in, const int* in_sizes, int n_in,
                              void* d_out, int out_size, void* d_ws,
                              size_t ws_size, hipStream_t stream) {
  (void)in_sizes; (void)n_in; (void)out_size; (void)ws_size;
  const float* query = (const float*)d_in[2];
  const float* W_qkv = (const float*)d_in[3];
  const float* b_qkv = (const float*)d_in[4];
  const float* W_out = (const float*)d_in[5];
  const float* b_out = (const float*)d_in[6];
  float* out = (float*)d_out;

  char* ws = (char*)d_ws;
  size_t off = 0;
  bf16_t* Xbf   = (bf16_t*)(ws + off); off += (size_t)M_ROWS * EMB * 2;
  bf16_t* Wqkvb = (bf16_t*)(ws + off); off += (size_t)N_QKV  * EMB * 2;
  bf16_t* Woutb = (bf16_t*)(ws + off); off += (size_t)EMB    * EMB * 2;
  bf16_t* Qb    = (bf16_t*)(ws + off); off += (size_t)BATCH * NH * S_LEN * HD * 2;
  bf16_t* Kb    = (bf16_t*)(ws + off); off += (size_t)BATCH * NH * S_LEN * HD * 2;
  bf16_t* Vt    = (bf16_t*)(ws + off); off += (size_t)BATCH * NH * S_LEN * HD * 2;
  bf16_t* Ctx   = (bf16_t*)(ws + off); off += (size_t)BATCH * NH * S_LEN * HD * 2;

  mha_cvt_bf16<<<1024, 256, 0, stream>>>(query, Xbf,   (M_ROWS * EMB) / 4);
  mha_cvt_bf16<<<1024, 256, 0, stream>>>(W_qkv, Wqkvb, (N_QKV * EMB) / 4);
  mha_cvt_bf16<<<512,  256, 0, stream>>>(W_out, Woutb, (EMB * EMB) / 4);

  dim3 g1(M_ROWS / 64, N_QKV / 64);
  mha_gemm_qkv<<<g1, 256, 0, stream>>>(Xbf, Wqkvb, b_qkv, Qb, Kb, Vt);

  const int nwaves = BATCH * NH * (S_LEN / 16);  // 4096
  mha_attn<<<nwaves / 8, 256, 0, stream>>>(Qb, Kb, Vt, Ctx);

  dim3 g3(M_ROWS / 64, EMB / 64);
  mha_gemm_out<<<g3, 256, 0, stream>>>(Ctx, Woutb, b_out, out);
}
